// RecallingModel_26929444946291
// MI455X (gfx1250) — matrix operand order repaired and checked by reading, not compiled, not run on hardware
//
#include <hip/hip_runtime.h>
#include <hip/hip_bf16.h>
#include <stdint.h>

// ---------------------------------------------------------------------------
// Problem constants (from the reference)
// ---------------------------------------------------------------------------
#define B_SZ   4096
#define H_SZ   512
#define A_SZ   18
#define D_SZ   1043          // 1024 + A + 1
#define M_SZ   32768
#define K_TOP  32
#define KC     1568          // D+H = 1555 padded to a multiple of 32
#define NG     2048          // 4*H
#define NEG_INF (-3.0e38f)

typedef __attribute__((ext_vector_type(16))) __bf16 v16bf;
typedef __attribute__((ext_vector_type(8)))  float  v8f;
typedef int v4i_t __attribute__((vector_size(16)));   // matches builtin param type

union Frag16 { v16bf v; __bf16 e[16]; uint4 q[2]; };

__device__ __forceinline__ float sigmoidf(float x) { return 1.0f / (1.0f + __expf(-x)); }

// ---------------------------------------------------------------------------
// Async global->LDS staging (CDNA5 GLOBAL_LOAD_ASYNC_TO_LDS_B128, ASYNCcnt),
// guarded so the file still compiles (with synchronous copies) if the builtin
// is absent on this toolchain. Prototype (from hipcc diagnostics):
//   void __builtin_amdgcn_global_load_async_to_lds_b128(
//       v4i_t AS1* src, v4i_t AS3* dst, int imm_offset, int cpol)
// ---------------------------------------------------------------------------
#if defined(__HIP_DEVICE_COMPILE__) && __has_builtin(__builtin_amdgcn_global_load_async_to_lds_b128)
#define USE_ASYNC_LDS 1
#else
#define USE_ASYNC_LDS 0
#endif

#if USE_ASYNC_LDS && __has_builtin(__builtin_amdgcn_s_wait_asynccnt)
#define WAIT_ASYNC(N) __builtin_amdgcn_s_wait_asynccnt(N)
#elif USE_ASYNC_LDS
#define WAIT_ASYNC(N) asm volatile("s_wait_asynccnt %0" ::"i"(N) : "memory")
#else
#define WAIT_ASYNC(N)
#endif

__device__ __forceinline__ void async16(const __bf16* g, __bf16* l) {
#if USE_ASYNC_LDS
  __builtin_amdgcn_global_load_async_to_lds_b128(
      (__attribute__((address_space(1))) v4i_t*)g,
      (__attribute__((address_space(3))) v4i_t*)l, 0, 0);
#else
  *(uint4*)l = *(const uint4*)g;
#endif
}

// ---------------------------------------------------------------------------
// Packing / conversion kernels (build bf16 operands for WMMA)
// ---------------------------------------------------------------------------
__global__ __launch_bounds__(256) void pack_a1_kernel(const float* __restrict__ x,
                                                      const float* __restrict__ h,
                                                      __bf16* __restrict__ A1) {
  size_t i = (size_t)blockIdx.x * blockDim.x + threadIdx.x;
  if (i >= (size_t)B_SZ * KC) return;
  int r = (int)(i / KC), col = (int)(i % KC);
  float v = 0.0f;
  if (col < D_SZ)              v = x[(size_t)r * D_SZ + col];
  else if (col < D_SZ + H_SZ)  v = h[(size_t)r * H_SZ + (col - D_SZ)];
  A1[i] = (__bf16)v;
}

// Pack combined LSTM weights TRANSPOSED: WcT[n][k], n-major. This makes the
// GEMM's B-tile staging a pure contiguous b128 copy (no in-kernel transpose,
// no scalar ds_store_b16), and B fragments stay contiguous b128 LDS loads.
__global__ __launch_bounds__(256) void pack_w_kernel(const float* __restrict__ wih,
                                                     const float* __restrict__ whh,
                                                     __bf16* __restrict__ WcT) {
  size_t i = (size_t)blockIdx.x * blockDim.x + threadIdx.x;
  if (i >= (size_t)KC * NG) return;
  int n = (int)(i / KC), k = (int)(i % KC);
  float v = 0.0f;
  if (k < D_SZ)              v = wih[(size_t)k * NG + n];
  else if (k < D_SZ + H_SZ)  v = whh[(size_t)(k - D_SZ) * NG + n];
  WcT[i] = (__bf16)v;
}

__global__ __launch_bounds__(256) void conv_bf16_kernel(const float* __restrict__ src,
                                                        __bf16* __restrict__ dst, size_t n) {
  size_t i = (size_t)blockIdx.x * blockDim.x + threadIdx.x;
  if (i < n) dst[i] = (__bf16)src[i];
}

// ---------------------------------------------------------------------------
// bf16 GEMM: C[f32, MxN] = A[bf16, MxK] * B, with B given TRANSPOSED (BT[N][K]).
// Block: 256 threads (8 waves), tile 128(M) x 64(N), K-step 32.
// Double-buffered async global->LDS staging: while WMMAs consume buffer `cur`,
// the next K-tile streams into `cur^1` (3 async b128 ops/thread, ASYNCcnt).
// ---------------------------------------------------------------------------
__global__ __launch_bounds__(256) void gemm_bf16_kernel(const __bf16* __restrict__ A,
                                                        const __bf16* __restrict__ BT,
                                                        float* __restrict__ C,
                                                        int Ndim, int Kdim) {
  __shared__ __bf16 ldsA[2][128][48];
  __shared__ __bf16 ldsB[2][64][48];

  const int tid  = threadIdx.x;
  const int wave = tid >> 5;
  const int lane = tid & 31;
  const int half = lane >> 4;
  const int m    = lane & 15;
  const int rowBase = blockIdx.y * 128;
  const int colBase = blockIdx.x * 64;

  // per-thread staging coordinates (16B chunks)
  const int ar0 = tid >> 2,         ak0 = (tid & 3) * 8;          // A chunks 0..255
  const int ar1 = (tid + 256) >> 2, ak1 = ((tid + 256) & 3) * 8;  // A chunks 256..511
  const int bn  = tid >> 2,         bk  = (tid & 3) * 8;          // B chunks 0..255

  v8f acc[4];
  #pragma unroll
  for (int t4 = 0; t4 < 4; ++t4)
    #pragma unroll
    for (int r = 0; r < 8; ++r) acc[t4][r] = 0.0f;

  const int nk = Kdim / 32;

  // prologue: stage K-tile 0 into buffer 0
  async16(A + (size_t)(rowBase + ar0) * Kdim + ak0, &ldsA[0][ar0][ak0]);
  async16(A + (size_t)(rowBase + ar1) * Kdim + ak1, &ldsA[0][ar1][ak1]);
  async16(BT + (size_t)(colBase + bn) * Kdim + bk,  &ldsB[0][bn][bk]);

  for (int s = 0; s < nk; ++s) {
    const int cur = s & 1;
    if (s + 1 < nk) {
      const int kb = (s + 1) * 32;
      async16(A + (size_t)(rowBase + ar0) * Kdim + kb + ak0, &ldsA[cur ^ 1][ar0][ak0]);
      async16(A + (size_t)(rowBase + ar1) * Kdim + kb + ak1, &ldsA[cur ^ 1][ar1][ak1]);
      async16(BT + (size_t)(colBase + bn) * Kdim + kb + bk,  &ldsB[cur ^ 1][bn][bk]);
      WAIT_ASYNC(3);   // previous stage's (in-order) 3 ops are complete
    } else {
      WAIT_ASYNC(0);
    }
    __syncthreads();

    Frag16 af;
    const __bf16* arow = &ldsA[cur][wave * 16 + m][0];
    af.q[0] = *(const uint4*)(arow + half * 16);
    af.q[1] = *(const uint4*)(arow + half * 16 + 8);

    Frag16 bfr[4];
    #pragma unroll
    for (int t4 = 0; t4 < 4; ++t4) {
      const __bf16* brow = &ldsB[cur][t4 * 16 + m][half * 16];
      bfr[t4].q[0] = ((const uint4*)brow)[0];
      bfr[t4].q[1] = ((const uint4*)brow)[1];
    }
    #pragma unroll
    for (int t4 = 0; t4 < 4; ++t4)
      acc[t4] = __builtin_amdgcn_wmma_f32_16x16x32_bf16(
          false, af.v, false, bfr[t4].v, (short)0, acc[t4], false, false);
    __syncthreads();
  }

  #pragma unroll
  for (int t4 = 0; t4 < 4; ++t4)
    #pragma unroll
    for (int r = 0; r < 8; ++r)
      C[(size_t)(rowBase + wave * 16 + r + half * 8) * Ndim + colBase + t4 * 16 + m] = acc[t4][r];
}

// ---------------------------------------------------------------------------
// LSTM elementwise: gates [B,2048] (f,i,o,g) + bias -> h,c.  Optionally packs
// h as bf16 into the next GEMM's A operand (cols D_SZ..D_SZ+H-1).
// ---------------------------------------------------------------------------
__global__ __launch_bounds__(256) void lstm_elem_kernel(const float* __restrict__ gates,
                                                        const float* __restrict__ bias,
                                                        const float* __restrict__ cprev,
                                                        float* __restrict__ hout,
                                                        float* __restrict__ cout,
                                                        __bf16* __restrict__ hpack) {
  size_t i = (size_t)blockIdx.x * blockDim.x + threadIdx.x;
  if (i >= (size_t)B_SZ * H_SZ) return;
  int r = (int)(i / H_SZ), j = (int)(i % H_SZ);
  const float* g = gates + (size_t)r * NG;
  float f  = g[j]            + bias[j];
  float ii = g[H_SZ + j]     + bias[H_SZ + j];
  float o  = g[2 * H_SZ + j] + bias[2 * H_SZ + j];
  float gg = g[3 * H_SZ + j] + bias[3 * H_SZ + j];
  float c = sigmoidf(f) * cprev[i] + sigmoidf(ii) * tanhf(gg);
  float h = sigmoidf(o) * tanhf(c);
  cout[i] = c;
  hout[i] = h;
  if (hpack) hpack[(size_t)r * KC + D_SZ + j] = (__bf16)h;
}

// ---------------------------------------------------------------------------
// LayerNorm(c1) -> q (bf16), one block (256 thr) per row of 512.
// ---------------------------------------------------------------------------
__global__ __launch_bounds__(256) void ln_q_kernel(const float* __restrict__ c1,
                                                   const float* __restrict__ g,
                                                   const float* __restrict__ b,
                                                   __bf16* __restrict__ q) {
  int r = blockIdx.x, t = threadIdx.x;
  __shared__ float red[256];
  const float* x = c1 + (size_t)r * H_SZ;
  float a0 = x[t], a1 = x[t + 256];
  red[t] = a0 + a1;
  __syncthreads();
  for (int off = 128; off; off >>= 1) { if (t < off) red[t] += red[t + off]; __syncthreads(); }
  float mean = red[0] * (1.0f / H_SZ);
  __syncthreads();
  float d0 = a0 - mean, d1 = a1 - mean;
  red[t] = d0 * d0 + d1 * d1;
  __syncthreads();
  for (int off = 128; off; off >>= 1) { if (t < off) red[t] += red[t + off]; __syncthreads(); }
  float inv = rsqrtf(red[0] * (1.0f / H_SZ) + 1e-5f);
  q[(size_t)r * H_SZ + t]       = (__bf16)(d0 * inv * g[t] + b[t]);
  q[(size_t)r * H_SZ + t + 256] = (__bf16)(d1 * inv * g[t + 256] + b[t + 256]);
}

// ---------------------------------------------------------------------------
// Fused scores + top-32 + softmax (block: 512 thr = 16 waves = 16 q-rows).
// Per 512-column chunk: each wave WMMAs two 16x16 score tiles (K=512, bf16,
// f32 acc; B fragments are contiguous 32B global reads from keys[M,H]), then
// wave w scans row w keeping a lane-distributed running top-32 (threshold +
// ballot -> ~220 expected insertions per row over 32768 cols). Finally
// sort-by-extraction + softmax.
// ---------------------------------------------------------------------------
#define TK_CHUNK 512
__global__ __launch_bounds__(512) void scores_topk_kernel(const __bf16* __restrict__ qb,
                                                          const __bf16* __restrict__ keys,
                                                          float* __restrict__ topw,
                                                          int* __restrict__ topi,
                                                          float* __restrict__ out_i,
                                                          float* __restrict__ out_w) {
  __shared__ __bf16 qtile[16][H_SZ];
  __shared__ float  sbuf[16][TK_CHUNK];
  __shared__ float  s_val[16][K_TOP];
  __shared__ int    s_idx[16][K_TOP];

  const int tid  = threadIdx.x;
  const int wave = tid >> 5;
  const int lane = tid & 31;
  const int half = lane >> 4;
  const int m    = lane & 15;
  const int row0 = blockIdx.x * 16;

  // stage the 16 q rows (16 KB = 1024 x 16B) into LDS, async if available
  {
    const __bf16* src = qb + (size_t)row0 * H_SZ;
    __bf16* dst = &qtile[0][0];
    for (int c = tid; c < 1024; c += 512) async16(src + c * 8, dst + c * 8);
    WAIT_ASYNC(0);
  }
  float rv = NEG_INF;   // running top-32, one entry per lane
  int   ri = -1;
  __syncthreads();

  for (int c = 0; c < M_SZ; c += TK_CHUNK) {
    // ---- compute 16 x TK_CHUNK scores with WMMA ----
    #pragma unroll
    for (int j = 0; j < 2; ++j) {
      int tn   = wave + 16 * j;
      int ncol = c + tn * 16 + m;                 // key row = score column
      const __bf16* brow = keys + (size_t)ncol * H_SZ;
      v8f acc;
      #pragma unroll
      for (int r = 0; r < 8; ++r) acc[r] = 0.0f;
      for (int ks = 0; ks < H_SZ; ks += 32) {
        Frag16 af, bfr;
        const __bf16* ar = &qtile[m][0] + ks;
        af.q[0] = *(const uint4*)(ar + half * 16);
        af.q[1] = *(const uint4*)(ar + half * 16 + 8);
        const __bf16* br = brow + ks + half * 16;
        bfr.q[0] = ((const uint4*)br)[0];
        bfr.q[1] = ((const uint4*)br)[1];
        acc = __builtin_amdgcn_wmma_f32_16x16x32_bf16(
            false, af.v, false, bfr.v, (short)0, acc, false, false);
      }
      #pragma unroll
      for (int r = 0; r < 8; ++r) sbuf[r + half * 8][tn * 16 + m] = acc[r];
    }
    __syncthreads();

    // ---- top-k update: wave w scans row w ----
    {
      const float* srow = &sbuf[wave][0];
      for (int b = 0; b < TK_CHUNK; b += 32) {
        float s   = srow[b + lane];
        int   sid = c + b + lane;
        float mn = rv;
        #pragma unroll
        for (int off = 16; off; off >>= 1) mn = fminf(mn, __shfl_xor(mn, off, 32));
        unsigned long long cand = __ballot(s > mn);
        while (cand) {
          int srcl = __ffsll((long long)cand) - 1;
          cand &= cand - 1;
          float cv = __shfl(s, srcl, 32);
          int   ci = __shfl(sid, srcl, 32);
          float mv = rv; int ml = lane;
          #pragma unroll
          for (int off = 16; off; off >>= 1) {
            float ov = __shfl_xor(mv, off, 32);
            int   ol = __shfl_xor(ml, off, 32);
            if (ov < mv || (ov == mv && ol < ml)) { mv = ov; ml = ol; }
          }
          if (cv > mv && lane == ml) { rv = cv; ri = ci; }
        }
      }
    }
    __syncthreads();
  }

  // ---- sort descending by repeated argmax extraction ----
  int taken = 0;
  for (int r = 0; r < K_TOP; ++r) {
    float av = taken ? NEG_INF : rv;
    int   ai = ri;
    int   al = lane;
    #pragma unroll
    for (int off = 16; off; off >>= 1) {
      float ov = __shfl_xor(av, off, 32);
      int   oi = __shfl_xor(ai, off, 32);
      int   ol = __shfl_xor(al, off, 32);
      if (ov > av || (ov == av && ol < al)) { av = ov; ai = oi; al = ol; }
    }
    if (lane == al) taken = 1;
    if (lane == 0) { s_val[wave][r] = av; s_idx[wave][r] = ai; }
  }
  __syncthreads();

  // ---- softmax over the sorted 32 and writeback ----
  float v  = s_val[wave][lane];
  float vm = s_val[wave][0];
  float e  = __expf(v - vm);
  float sum = e;
  #pragma unroll
  for (int off = 16; off; off >>= 1) sum += __shfl_xor(sum, off, 32);
  float w = e / sum;
  int row = row0 + wave;
  int id  = s_idx[wave][lane];
  topw[(size_t)row * K_TOP + lane]  = w;
  topi[(size_t)row * K_TOP + lane]  = id;
  out_w[(size_t)row * K_TOP + lane] = w;
  out_i[(size_t)row * K_TOP + lane] = (float)id;
}

// ---------------------------------------------------------------------------
// memories[b,:] = sum_k w[b,k] * mem_values[idx[b,k],:]  -> bf16 into A2
// ---------------------------------------------------------------------------
__global__ __launch_bounds__(256) void gather_kernel(const float* __restrict__ w,
                                                     const int* __restrict__ idx,
                                                     const float* __restrict__ mv,
                                                     __bf16* __restrict__ A2) {
  int r = blockIdx.x;
  __shared__ float wts[K_TOP];
  __shared__ int   ids[K_TOP];
  if (threadIdx.x < K_TOP) {
    wts[threadIdx.x] = w[(size_t)r * K_TOP + threadIdx.x];
    ids[threadIdx.x] = idx[(size_t)r * K_TOP + threadIdx.x];
  }
  __syncthreads();
  for (int col = threadIdx.x; col < D_SZ; col += 256) {
    if (col + 256 < D_SZ) __builtin_prefetch(mv + (size_t)ids[0] * D_SZ + col + 256, 0, 1);
    float acc = 0.0f;
    #pragma unroll 8
    for (int k = 0; k < K_TOP; ++k) acc += wts[k] * mv[(size_t)ids[k] * D_SZ + col];
    A2[(size_t)r * KC + col] = (__bf16)acc;
  }
}

// ---------------------------------------------------------------------------
// critic / actor heads: one wave per row, shfl-reduced dot products (tiny N).
// ---------------------------------------------------------------------------
__global__ __launch_bounds__(256) void head_kernel(const float* __restrict__ h2,
                                                   const float* __restrict__ Wc,
                                                   const float* __restrict__ bc,
                                                   const float* __restrict__ Wa,
                                                   const float* __restrict__ ba,
                                                   float* __restrict__ critic,
                                                   float* __restrict__ actor) {
  int wave = threadIdx.x >> 5, lane = threadIdx.x & 31;
  int r = blockIdx.x * 8 + wave;
  const float* hr = h2 + (size_t)r * H_SZ;
  float hv[16];
  #pragma unroll
  for (int i = 0; i < 16; ++i) hv[i] = hr[lane + 32 * i];
  float s = 0.0f;
  #pragma unroll
  for (int i = 0; i < 16; ++i) s += hv[i] * Wc[lane + 32 * i];
  #pragma unroll
  for (int off = 16; off; off >>= 1) s += __shfl_xor(s, off, 32);
  if (lane == 0) critic[r] = s + bc[0];
  for (int a = 0; a < A_SZ; ++a) {
    float t = 0.0f;
    #pragma unroll
    for (int i = 0; i < 16; ++i) t += hv[i] * Wa[(size_t)(lane + 32 * i) * A_SZ + a];
    #pragma unroll
    for (int off = 16; off; off >>= 1) t += __shfl_xor(t, off, 32);
    if (lane == 0) actor[(size_t)r * A_SZ + a] = t + ba[a];
  }
}

// ---------------------------------------------------------------------------
// Host-side orchestration
// ---------------------------------------------------------------------------
static inline size_t alignup(size_t x) { return (x + 255) & ~(size_t)255; }

extern "C" void kernel_launch(void* const* d_in, const int* in_sizes, int n_in,
                              void* d_out, int out_size, void* d_ws, size_t ws_size,
                              hipStream_t stream) {
  (void)in_sizes; (void)n_in; (void)out_size; (void)ws_size;
  const float* env      = (const float*)d_in[0];
  const float* hxs      = (const float*)d_in[1];
  const float* cxs      = (const float*)d_in[2];
  const float* mem_keys = (const float*)d_in[3];
  const float* mem_vals = (const float*)d_in[4];
  const float* W_ih     = (const float*)d_in[5];
  const float* W_hh     = (const float*)d_in[6];
  const float* bias     = (const float*)d_in[7];
  const float* ln_g     = (const float*)d_in[8];
  const float* ln_b     = (const float*)d_in[9];
  const float* W_crit   = (const float*)d_in[10];
  const float* b_crit   = (const float*)d_in[11];
  const float* W_act    = (const float*)d_in[12];
  const float* b_act    = (const float*)d_in[13];

  // workspace carve-up
  char* w = (char*)d_ws;
  __bf16* A1    = (__bf16*)w; w += alignup((size_t)B_SZ * KC * 2);
  __bf16* WcT   = (__bf16*)w; w += alignup((size_t)KC * NG * 2);
  __bf16* Keysb = (__bf16*)w; w += alignup((size_t)M_SZ * H_SZ * 2);
  float*  Gates = (float*)w;  w += alignup((size_t)B_SZ * NG * 4);
  float*  H1    = (float*)w;  w += alignup((size_t)B_SZ * H_SZ * 4);
  float*  C1    = (float*)w;  w += alignup((size_t)B_SZ * H_SZ * 4);
  __bf16* Qb    = (__bf16*)w; w += alignup((size_t)B_SZ * H_SZ * 2);
  __bf16* A2    = (__bf16*)w; w += alignup((size_t)B_SZ * KC * 2);
  float*  TopW  = (float*)w;  w += alignup((size_t)B_SZ * K_TOP * 4);
  int*    TopI  = (int*)w;    w += alignup((size_t)B_SZ * K_TOP * 4);

  // output carve-up (return order: critic, actor, h2, c2, inds, weights)
  float* out      = (float*)d_out;
  float* o_critic = out;
  float* o_actor  = o_critic + (size_t)B_SZ;
  float* o_h2     = o_actor  + (size_t)B_SZ * A_SZ;
  float* o_c2     = o_h2     + (size_t)B_SZ * H_SZ;
  float* o_inds   = o_c2     + (size_t)B_SZ * H_SZ;
  float* o_wts    = o_inds   + (size_t)B_SZ * K_TOP;

  // A2 must have zero pad columns (1555..1567); gather/lstm fill the rest.
  (void)hipMemsetAsync(A2, 0, (size_t)B_SZ * KC * 2, stream);

  pack_a1_kernel<<<(B_SZ * KC) / 256, 256, 0, stream>>>(env, hxs, A1);
  pack_w_kernel<<<(KC * NG) / 256, 256, 0, stream>>>(W_ih, W_hh, WcT);
  conv_bf16_kernel<<<(M_SZ * H_SZ) / 256, 256, 0, stream>>>(mem_keys, Keysb, (size_t)M_SZ * H_SZ);

  // LSTM pass 1
  gemm_bf16_kernel<<<dim3(NG / 64, B_SZ / 128), 256, 0, stream>>>(A1, WcT, Gates, NG, KC);
  lstm_elem_kernel<<<(B_SZ * H_SZ) / 256, 256, 0, stream>>>(Gates, bias, cxs, H1, C1, A2);

  // retrieval
  ln_q_kernel<<<B_SZ, 256, 0, stream>>>(C1, ln_g, ln_b, Qb);
  scores_topk_kernel<<<B_SZ / 16, 512, 0, stream>>>(Qb, Keysb, TopW, TopI, o_inds, o_wts);
  gather_kernel<<<B_SZ, 256, 0, stream>>>(TopW, TopI, mem_vals, A2);

  // LSTM pass 2 (h2/c2 straight into d_out)
  gemm_bf16_kernel<<<dim3(NG / 64, B_SZ / 128), 256, 0, stream>>>(A2, WcT, Gates, NG, KC);
  lstm_elem_kernel<<<(B_SZ * H_SZ) / 256, 256, 0, stream>>>(Gates, bias, C1, o_h2, o_c2, nullptr);

  // heads
  head_kernel<<<B_SZ / 8, 256, 0, stream>>>(o_h2, W_crit, b_crit, W_act, b_act, o_critic, o_actor);
}
